// gMLP_54357106098474
// MI455X (gfx1250) — matrix operand order repaired and checked by reading, not compiled, not run on hardware
//
#include <hip/hip_runtime.h>
#include <math.h>

// ---------------- problem constants (match reference) ----------------
#define NN   50000
#define EE   400000
#define FIN  128
#define HH   128
#define FFD  256
#define LL   3
#define CC   40
#define CPAD 48      // pad classifier cols 40 -> 48 (3 x 16 tiles)

typedef __attribute__((ext_vector_type(16))) __bf16 v16bf;
typedef __attribute__((ext_vector_type(8)))  __bf16 v8bf;
typedef __attribute__((ext_vector_type(8)))  float  v8f;

__device__ __forceinline__ __bf16 f2bf(float f) {
  unsigned u = __builtin_bit_cast(unsigned, f);
  u += 0x7fffu + ((u >> 16) & 1u);            // round-to-nearest-even
  unsigned short h = (unsigned short)(u >> 16);
  return __builtin_bit_cast(__bf16, h);
}

// ---------------- weight fp32 [K][Nc] -> bf16 transposed [NcPad][K] ----------------
__global__ void cvt_w_t_kernel(const float* __restrict__ s, __bf16* __restrict__ d,
                               int K, int Nc, int NcPad) {
  int total = NcPad * K;
  for (int i = blockIdx.x * blockDim.x + threadIdx.x; i < total; i += gridDim.x * blockDim.x) {
    int n = i / K, k = i % K;
    d[i] = (n < Nc) ? f2bf(s[k * Nc + n]) : f2bf(0.0f);
  }
}

// ---------------- activation fp32 -> bf16 staging ----------------
__global__ void cvt_act_kernel(const float* __restrict__ s, __bf16* __restrict__ d, int n) {
  for (int i = blockIdx.x * blockDim.x + threadIdx.x; i < n; i += gridDim.x * blockDim.x)
    d[i] = f2bf(s[i]);
}

// ---------------- LayerNorm (block per row, D = blockDim.x), bf16 out ----------------
__global__ void ln_kernel(const float* __restrict__ in, __bf16* __restrict__ out,
                          const float* __restrict__ g, const float* __restrict__ b) {
  __shared__ float red[256];
  const int D = blockDim.x;
  const int row = blockIdx.x, tid = threadIdx.x;
  float v = in[row * D + tid];
  red[tid] = v; __syncthreads();
  for (int s = D >> 1; s > 0; s >>= 1) { if (tid < s) red[tid] += red[tid + s]; __syncthreads(); }
  float mean = red[0] / (float)D; __syncthreads();
  float dv = v - mean;
  red[tid] = dv * dv; __syncthreads();
  for (int s = D >> 1; s > 0; s >>= 1) { if (tid < s) red[tid] += red[tid + s]; __syncthreads(); }
  float var = red[0] / (float)D;
  out[row * D + tid] = f2bf(dv * rsqrtf(var + 1e-5f) * g[tid] + b[tid]);
}

// ---------------- GCN degree / normalization ----------------
__global__ void deg_init_kernel(float* __restrict__ deg) {
  int i = blockIdx.x * blockDim.x + threadIdx.x;
  if (i < NN) deg[i] = 2.0f;                 // improved=True self-loop weight
}
__global__ void deg_edges_kernel(const long long* __restrict__ ei, float* __restrict__ deg) {
  int e = blockIdx.x * blockDim.x + threadIdx.x;
  if (e < EE) atomicAdd(&deg[(int)ei[EE + e]], 1.0f);   // segment over col
}
__global__ void deg_inv_kernel(float* __restrict__ deg) {
  int i = blockIdx.x * blockDim.x + threadIdx.x;
  if (i < NN) { float d = deg[i]; deg[i] = (d > 0.0f) ? rsqrtf(d) : 0.0f; }
}

// ---------------- GCN aggregation ----------------
// self-loop: full overwrite of agg (also re-initializes poisoned workspace)
__global__ void gcn_self_kernel(const float* __restrict__ g1, const float* __restrict__ dinv,
                                float* __restrict__ agg) {
  int total = NN * FFD;
  for (int i = blockIdx.x * blockDim.x + threadIdx.x; i < total; i += gridDim.x * blockDim.x) {
    int node = i >> 8;                       // FFD == 256
    float di = dinv[node];
    agg[i] = 2.0f * di * di * g1[i];
  }
}

__global__ void gcn_scatter_kernel(const long long* __restrict__ ei, const float* __restrict__ dinv,
                                   const float* __restrict__ g1, float* __restrict__ agg) {
  const long long total = (long long)EE * (FFD / 4);
  for (long long idx = blockIdx.x * (long long)blockDim.x + threadIdx.x; idx < total;
       idx += (long long)gridDim.x * blockDim.x) {
    int e = (int)(idx >> 6);                 // FFD/4 == 64 chunks per edge
    int c = ((int)idx & 63) << 2;
    int r  = (int)ei[e];
    int cl = (int)ei[EE + e];
    float w = dinv[r] * dinv[cl];            // ew == 1 for real edges
    const float* src = g1 + (size_t)r * FFD + c;
    float* dst = agg + (size_t)cl * FFD + c;
    atomicAdd(dst + 0, w * src[0]);
    atomicAdd(dst + 1, w * src[1]);
    atomicAdd(dst + 2, w * src[2]);
    atomicAdd(dst + 3, w * src[3]);
  }
}

// gate = tanh(agg + b_gcn); out_bf16 = t * gate  (t itself stays untouched)
__global__ void gate_apply_kernel(const float* __restrict__ t, const float* __restrict__ agg,
                                  const float* __restrict__ bg, __bf16* __restrict__ out) {
  int total = NN * FFD;
  for (int i = blockIdx.x * blockDim.x + threadIdx.x; i < total; i += gridDim.x * blockDim.x)
    out[i] = f2bf(t[i] * tanhf(agg[i] + bg[i & 255]));
}

// ---------------- WMMA GEMM ----------------
// C[M,Nc] = act(A[M,K]_bf16 @ B[K,Nc] + bias) (+ res), B stored transposed bf16 [NcPad][K].
// One wave computes 16 rows x (CT*16) cols; 8 waves / 256-thread block.
// Fragments are pairs of contiguous 16B loads matching the ISA 7.12.2 bf16 layouts:
//   A row-major:  lane(l15,hi) loads K = k0+8*hi..+7 and k0+16+8*hi..+7 of row rt*16+l15
//   B transposed: lane(l15,hi) loads K = k0+8*hi..+7 and k0+16+8*hi..+7 of column nc+l15
template <int K, int CT>
__global__ void wmma_gemm(const __bf16* __restrict__ A, const __bf16* __restrict__ Bt,
                          const float* __restrict__ bias, const float* __restrict__ res,
                          float* __restrict__ C, int ldout, int nvalid, int act) {
  const int wave = threadIdx.x >> 5;
  const int lane = threadIdx.x & 31;
  const int rt = blockIdx.x * 8 + wave;               // wave-uniform
  if (rt * 16 >= NN) return;
  const int nc0 = blockIdx.y * (CT * 16);
  const int l15 = lane & 15;
  const int hi  = lane >> 4;
  const __bf16* arow = A + (size_t)(rt * 16 + l15) * K;

  v8f acc[CT];
#pragma unroll
  for (int c = 0; c < CT; ++c) acc[c] = (v8f){0.f, 0.f, 0.f, 0.f, 0.f, 0.f, 0.f, 0.f};

#pragma unroll
  for (int k0 = 0; k0 < K; k0 += 32) {
    const v8bf a0 = *(const v8bf*)(arow + k0 + hi * 8);
    const v8bf a1 = *(const v8bf*)(arow + k0 + 16 + hi * 8);
    const v16bf afrag = __builtin_shufflevector(a0, a1, 0, 1, 2, 3, 4, 5, 6, 7,
                                                8, 9, 10, 11, 12, 13, 14, 15);
#pragma unroll
    for (int c = 0; c < CT; ++c) {
      const __bf16* bcol = Bt + (size_t)(nc0 + c * 16 + l15) * K + k0 + hi * 8;
      const v8bf b0 = *(const v8bf*)(bcol);
      const v8bf b1 = *(const v8bf*)(bcol + 16);
      const v16bf bfrag = __builtin_shufflevector(b0, b1, 0, 1, 2, 3, 4, 5, 6, 7,
                                                  8, 9, 10, 11, 12, 13, 14, 15);
      acc[c] = __builtin_amdgcn_wmma_f32_16x16x32_bf16(false, afrag, false, bfrag,
                                                       (short)0, acc[c], false, false);
    }
  }

#pragma unroll
  for (int c = 0; c < CT; ++c) {
    const int col = nc0 + c * 16 + l15;
    if (col < nvalid) {
#pragma unroll
      for (int i = 0; i < 8; ++i) {
        const int m = rt * 16 + i + (hi << 3);        // C/D layout: VGPR i -> M=i (+8 hi half)
        float val = acc[c][i];
        if (bias) val += bias[col];
        if (act == 1) val = 0.5f * val * (1.0f + erff(val * 0.70710678118654752f));
        if (res)  val += res[m * ldout + col];
        C[m * ldout + col] = val;
      }
    }
  }
}

// ---------------- host launch ----------------
extern "C" void kernel_launch(void* const* d_in, const int* in_sizes, int n_in,
                              void* d_out, int out_size, void* d_ws, size_t ws_size,
                              hipStream_t stream) {
  (void)in_sizes; (void)n_in; (void)out_size; (void)ws_size;

  const float*      x      = (const float*)d_in[0];
  const long long*  ei     = (const long long*)d_in[1];   // int64 edge_index [2,E]
  const float*      W_emb  = (const float*)d_in[2];
  const float*      b_emb  = (const float*)d_in[3];
  const float*      ln1_g  = (const float*)d_in[4];
  const float*      ln1_b  = (const float*)d_in[5];
  const float*      W_in   = (const float*)d_in[6];
  const float*      b_in   = (const float*)d_in[7];
  const float*      ln2_g  = (const float*)d_in[8];
  const float*      ln2_b  = (const float*)d_in[9];
  const float*      W_gcn  = (const float*)d_in[10];
  const float*      b_gcn  = (const float*)d_in[11];
  const float*      W_out  = (const float*)d_in[12];
  const float*      b_out  = (const float*)d_in[13];
  const float*      W_cls  = (const float*)d_in[14];
  const float*      b_cls  = (const float*)d_in[15];
  float*            out    = (float*)d_out;

  // workspace carve-out
  char* ws = (char*)d_ws;
  size_t off = 0;
  auto take = [&](size_t bytes) -> void* {
    void* p = ws + off;
    off += bytes;
    off = (off + 255) & ~(size_t)255;
    return p;
  };
  float*  h    = (float*)take((size_t)NN * HH  * 4);
  float*  t    = (float*)take((size_t)NN * FFD * 4);
  float*  g1   = (float*)take((size_t)NN * FFD * 4);   // dense gcn output
  float*  agg  = (float*)take((size_t)NN * FFD * 4);   // aggregation result
  float*  dinv = (float*)take((size_t)NN * 4);
  __bf16* abf  = (__bf16*)take((size_t)NN * FFD * 2);  // shared bf16 GEMM-A staging
  __bf16* wemb_bf = (__bf16*)take((size_t)HH * FIN * 2);        // [128][128] transposed
  __bf16* win_bf  = (__bf16*)take((size_t)LL * FFD * HH * 2);   // [256][128] per layer
  __bf16* wgcn_bf = (__bf16*)take((size_t)LL * FFD * FFD * 2);  // [256][256] per layer
  __bf16* wout_bf = (__bf16*)take((size_t)LL * HH * FFD * 2);   // [128][256] per layer
  __bf16* wcls_bf = (__bf16*)take((size_t)CPAD * HH * 2);       // [48][128], rows 40..47 zero

  const dim3 blk(256, 1, 1);

  // --- weight conversion + transpose (every call: deterministic, replay-safe) ---
  {
    int n;
    n = HH * FIN;
    cvt_w_t_kernel<<<(n + 255) / 256, blk, 0, stream>>>(W_emb, wemb_bf, FIN, HH, HH);
    for (int l = 0; l < LL; ++l) {
      n = FFD * HH;
      cvt_w_t_kernel<<<(n + 255) / 256, blk, 0, stream>>>(W_in + (size_t)l * HH * FFD,
                                                          win_bf + (size_t)l * FFD * HH,
                                                          HH, FFD, FFD);
      n = FFD * FFD;
      cvt_w_t_kernel<<<(n + 255) / 256, blk, 0, stream>>>(W_gcn + (size_t)l * FFD * FFD,
                                                          wgcn_bf + (size_t)l * FFD * FFD,
                                                          FFD, FFD, FFD);
      n = HH * FFD;
      cvt_w_t_kernel<<<(n + 255) / 256, blk, 0, stream>>>(W_out + (size_t)l * FFD * HH,
                                                          wout_bf + (size_t)l * HH * FFD,
                                                          FFD, HH, HH);
    }
    n = CPAD * HH;
    cvt_w_t_kernel<<<(n + 255) / 256, blk, 0, stream>>>(W_cls, wcls_bf, HH, CC, CPAD);
  }

  // --- degree normalization (once, shared by all layers) ---
  deg_init_kernel<<<(NN + 255) / 256, blk, 0, stream>>>(dinv);
  deg_edges_kernel<<<(EE + 255) / 256, blk, 0, stream>>>(ei, dinv);
  deg_inv_kernel<<<(NN + 255) / 256, blk, 0, stream>>>(dinv);

  const int row_blocks = (NN / 16 + 7) / 8;            // 3125 tiles -> 391 blocks
  const int NF = NN * FFD;
  const int nf_blocks = (NF + 255) / 256;
  const int nh_blocks = (NN * HH + 255) / 256;
  const long long scat_total = (long long)EE * (FFD / 4);
  const int scat_blocks = (int)((scat_total + 255) / 256);

  // h = x @ W_emb + b_emb
  cvt_act_kernel<<<nh_blocks, blk, 0, stream>>>(x, abf, NN * FIN);
  wmma_gemm<128, 4><<<dim3(row_blocks, HH / 64), blk, 0, stream>>>(
      abf, wemb_bf, b_emb, nullptr, h, HH, HH, 0);

  for (int l = 0; l < LL; ++l) {
    // t = gelu(LN1(h) @ W_in + b_in)
    ln_kernel<<<NN, HH, 0, stream>>>(h, abf, ln1_g + l * HH, ln1_b + l * HH);
    wmma_gemm<128, 4><<<dim3(row_blocks, FFD / 64), blk, 0, stream>>>(
        abf, win_bf + (size_t)l * FFD * HH, b_in + l * FFD, nullptr, t, FFD, FFD, 1);

    // gate = LN2(t); g1 = gate @ W_gcn
    ln_kernel<<<NN, FFD, 0, stream>>>(t, abf, ln2_g + l * FFD, ln2_b + l * FFD);
    wmma_gemm<256, 4><<<dim3(row_blocks, FFD / 64), blk, 0, stream>>>(
        abf, wgcn_bf + (size_t)l * FFD * FFD, nullptr, nullptr, g1, FFD, FFD, 0);

    // agg = D^-1/2 A_hat D^-1/2 @ g1  (self-loop weight 2, symmetric norm)
    gcn_self_kernel<<<nf_blocks, blk, 0, stream>>>(g1, dinv, agg);
    gcn_scatter_kernel<<<scat_blocks, blk, 0, stream>>>(ei, dinv, g1, agg);

    // abf = bf16( t * tanh(agg + b_gcn) )
    gate_apply_kernel<<<nf_blocks, blk, 0, stream>>>(t, agg, b_gcn + l * FFD, abf);

    // h = h + (gated t) @ W_out + b_out  (residual fused; 1 read + 1 write per element)
    wmma_gemm<256, 4><<<dim3(row_blocks, HH / 64), blk, 0, stream>>>(
        abf, wout_bf + (size_t)l * HH * FFD, b_out + l * HH, h, h, HH, HH, 0);
  }

  // out = h @ W_cls + b_cls   (cols padded to 48, clipped at 40, ld_out = 40)
  cvt_act_kernel<<<nh_blocks, blk, 0, stream>>>(h, abf, NN * HH);
  wmma_gemm<128, 3><<<dim3(row_blocks, 1), blk, 0, stream>>>(
      abf, wcls_bf, b_cls, nullptr, out, CC, CC, 0);
}
